// EGCNII_1374389534970
// MI455X (gfx1250) — matrix operand order repaired and checked by reading, not compiled, not run on hardware
//
#include <hip/hip_runtime.h>
#include <cmath>

typedef __attribute__((ext_vector_type(16))) _Float16 v16h;
typedef __attribute__((ext_vector_type(8)))  float    v8f;

#define ONE_MINUS_ALPHA 0.9f
#define ALPHA_C 0.1f
#define LNEPS 1e-5f

// K index held by half-index v (0..15) of a v16h A-operand, for lane-half `half`.
// ISA 16-bit A 16x32 layout: VGPR p: low half K=2p,2p+1 (+16 for p>=4), high half +8.
__device__ __forceinline__ int a_k(int v, int half) {
  int p = v >> 1;
  return ((p >> 2) << 4) + (half << 3) + ((p & 3) << 1) + (v & 1);
}

__device__ __forceinline__ v8f wmma_f16(v16h a, v16h b, v8f c) {
  return __builtin_amdgcn_wmma_f32_16x16x32_f16(false, a, false, b, (short)0, c,
                                                false, false);
}

// ---------------- small utility kernels ----------------

__global__ void f32_to_f16_kernel(const float* __restrict__ src,
                                  _Float16* __restrict__ dst, int n) {
  int i = blockIdx.x * blockDim.x + threadIdx.x;
  if (i < n) dst[i] = (_Float16)src[i];
}

__global__ void deg_kernel(const int* __restrict__ col, float* __restrict__ deg, int E) {
  int e = blockIdx.x * blockDim.x + threadIdx.x;
  if (e < E) atomicAdd(&deg[col[e]], 1.0f);
}

__global__ void dinv_kernel(float* __restrict__ deg, int N) {
  int i = blockIdx.x * blockDim.x + threadIdx.x;
  if (i < N) deg[i] = rsqrtf(deg[i] + 1.0f);  // +1 = self loop; always > 0
}

// ---------------- input linear: h = relu(x @ W + b); h0 = h ----------------

__global__ void lin_kernel(const float* __restrict__ x, const _Float16* __restrict__ W,
                           const float* __restrict__ bias, float* __restrict__ h,
                           float* __restrict__ h0, int N) {
  int wave = blockIdx.x * (blockDim.x >> 5) + (threadIdx.x >> 5);
  int lane = threadIdx.x & 31;
  int m0 = wave << 4;
  if (m0 >= N) return;
  int half = lane >> 4, mrow = lane & 15;
  int node = m0 + mrow; if (node > N - 1) node = N - 1;
  const float* xr = x + (size_t)node * 64;

  v16h a0, a1;
#pragma unroll
  for (int v = 0; v < 16; ++v) {
    int k = a_k(v, half);
    a0[v] = (_Float16)xr[k];
    a1[v] = (_Float16)xr[k + 32];
  }
  v8f c[4];
#pragma unroll
  for (int t = 0; t < 4; ++t) c[t] = 0.0f;
#pragma unroll
  for (int t = 0; t < 4; ++t) {
    v16h b; for (int v = 0; v < 16; ++v) b[v] = W[(size_t)a_k(v, half) * 64 + (t << 4) + mrow];     // K = a_k(v, half)
    c[t] = wmma_f16(a0, b, c[t]);
  }
#pragma unroll
  for (int t = 0; t < 4; ++t) {
    v16h b; for (int v = 0; v < 16; ++v) b[v] = W[(size_t)(32 + a_k(v, half)) * 64 + (t << 4) + mrow]; // K = 32+a_k(v, half)
    c[t] = wmma_f16(a1, b, c[t]);
  }
#pragma unroll
  for (int t = 0; t < 4; ++t)
#pragma unroll
    for (int r = 0; r < 8; ++r) {
      int mm = m0 + r + (half << 3);
      int nn = (t << 4) + mrow;
      if (mm < N) {
        float v = fmaxf(c[t][r] + bias[nn], 0.0f);
        h[(size_t)mm * 64 + nn]  = v;
        h0[(size_t)mm * 64 + nn] = v;
      }
    }
}

// ---------------- SpMM scatter: agg[row] += dinv[row]*dinv[col]*h[col] ----------------

__global__ void spmm_kernel(const float* __restrict__ h, const int* __restrict__ row,
                            const int* __restrict__ col, const float* __restrict__ dinv,
                            float* __restrict__ agg, int E) {
  int t = blockIdx.x * blockDim.x + threadIdx.x;
  int e = t >> 4;                 // 16 threads per edge, 4 feats each
  if (e >= E) return;
  int f = (t & 15) << 2;
  int r = row[e], c = col[e];
  float w = dinv[r] * dinv[c];
  const float4 hv = *(const float4*)(h + (size_t)c * 64 + f);
  float* ar = agg + (size_t)r * 64 + f;
  atomicAdd(ar + 0, w * hv.x);
  atomicAdd(ar + 1, w * hv.y);
  atomicAdd(ar + 2, w * hv.z);
  atomicAdd(ar + 3, w * hv.w);
}

// ---------------- GCNII layer: h' = relu((1-b)*mix + b*(mix@W)) ----------------
// mix = 0.9*(agg + dinv^2*h) + 0.1*h0   (self loop folded here)

__global__ void layer_kernel(const float* __restrict__ agg, const float* __restrict__ h,
                             const float* __restrict__ h0, const float* __restrict__ dinv,
                             const _Float16* __restrict__ W, float* __restrict__ hn,
                             float beta, int N) {
  int wave = blockIdx.x * (blockDim.x >> 5) + (threadIdx.x >> 5);
  int lane = threadIdx.x & 31;
  int m0 = wave << 4;
  if (m0 >= N) return;
  int half = lane >> 4, mrow = lane & 15;
  int node = m0 + mrow; if (node > N - 1) node = N - 1;
  float di = dinv[node];
  float slc = ONE_MINUS_ALPHA * di * di;
  const float* ar  = agg + (size_t)node * 64;
  const float* hr  = h   + (size_t)node * 64;
  const float* h0r = h0  + (size_t)node * 64;

  v16h a0, a1;
#pragma unroll
  for (int v = 0; v < 16; ++v) {
    int k = a_k(v, half);
    a0[v] = (_Float16)(ONE_MINUS_ALPHA * ar[k] + slc * hr[k] + ALPHA_C * h0r[k]);
    int k2 = k + 32;
    a1[v] = (_Float16)(ONE_MINUS_ALPHA * ar[k2] + slc * hr[k2] + ALPHA_C * h0r[k2]);
  }
  v8f c[4];
#pragma unroll
  for (int t = 0; t < 4; ++t) c[t] = 0.0f;
#pragma unroll
  for (int t = 0; t < 4; ++t) {
    v16h b; for (int v = 0; v < 16; ++v) b[v] = W[(size_t)a_k(v, half) * 64 + (t << 4) + mrow];
    c[t] = wmma_f16(a0, b, c[t]);
  }
#pragma unroll
  for (int t = 0; t < 4; ++t) {
    v16h b; for (int v = 0; v < 16; ++v) b[v] = W[(size_t)(32 + a_k(v, half)) * 64 + (t << 4) + mrow];
    c[t] = wmma_f16(a1, b, c[t]);
  }
#pragma unroll
  for (int t = 0; t < 4; ++t)
#pragma unroll
    for (int r = 0; r < 8; ++r) {
      int mm = m0 + r + (half << 3);
      int nn = (t << 4) + mrow;
      if (mm < N) {
        float d2 = dinv[mm];
        size_t idx = (size_t)mm * 64 + nn;
        float mix = ONE_MINUS_ALPHA * agg[idx] +
                    ONE_MINUS_ALPHA * d2 * d2 * h[idx] + ALPHA_C * h0[idx];
        float val = (1.0f - beta) * mix + beta * c[t][r];
        hn[idx] = fmaxf(val, 0.0f);
      }
    }
}

// ---------------- fused classifier over edges ----------------
// per 16-edge tile: gather [h[src]|h[dst]] (16x128) -> WMMA -> +b1 -> LayerNorm
// (half-wave shfl_xor reductions) -> ReLU -> 64x10 GEMM via LDS -> out[E,10]

__global__ void cls_kernel(const float* __restrict__ h, const int* __restrict__ src,
                           const int* __restrict__ dst, const _Float16* __restrict__ W1,
                           const float* __restrict__ b1, const float* __restrict__ lng,
                           const float* __restrict__ lnb, const float* __restrict__ W2,
                           const float* __restrict__ b2, float* __restrict__ out, int E) {
  __shared__ float zsh[8][16 * 64];   // per-wave z tile, 32 KB total
  int wv   = threadIdx.x >> 5;
  int wave = blockIdx.x * (blockDim.x >> 5) + wv;
  int lane = threadIdx.x & 31;
  int e0 = wave << 4;
  int half = lane >> 4, mrow = lane & 15;
  int edge = e0 + mrow; if (edge > E - 1) edge = E - 1;
  const float* hs = h + (size_t)src[edge] * 64;
  const float* hd = h + (size_t)dst[edge] * 64;

  v16h a[4];
#pragma unroll
  for (int gI = 0; gI < 4; ++gI)
#pragma unroll
    for (int v = 0; v < 16; ++v) {
      int k = (gI << 5) + a_k(v, half);
      a[gI][v] = (_Float16)((k < 64) ? hs[k] : hd[k - 64]);
    }
  v8f c[4];
#pragma unroll
  for (int t = 0; t < 4; ++t) c[t] = 0.0f;
#pragma unroll
  for (int gI = 0; gI < 4; ++gI) {
    int kr = (gI << 5);                 // B row base of this K step
#pragma unroll
    for (int t = 0; t < 4; ++t) {
      v16h b; for (int v = 0; v < 16; ++v) b[v] = W1[(size_t)(kr + a_k(v, half)) * 64 + (t << 4) + mrow];
      c[t] = wmma_f16(a[gI], b, c[t]);
    }
  }
  // bias (N index = 16t + mrow in C layout)
#pragma unroll
  for (int t = 0; t < 4; ++t)
#pragma unroll
    for (int r = 0; r < 8; ++r) c[t][r] += b1[(t << 4) + mrow];

  // LayerNorm per edge row: row (r + 8*half) lives across 16 lanes of a half-wave
#pragma unroll
  for (int r = 0; r < 8; ++r) {
    float s1 = c[0][r] + c[1][r] + c[2][r] + c[3][r];
    float s2 = c[0][r]*c[0][r] + c[1][r]*c[1][r] + c[2][r]*c[2][r] + c[3][r]*c[3][r];
#pragma unroll
    for (int off = 8; off; off >>= 1) {
      s1 += __shfl_xor(s1, off, 16);
      s2 += __shfl_xor(s2, off, 16);
    }
    float mu  = s1 * 0.015625f;
    float var = s2 * 0.015625f - mu * mu;
    float rs  = rsqrtf(var + LNEPS);
#pragma unroll
    for (int t = 0; t < 4; ++t) {
      int nn = (t << 4) + mrow;
      float z = (c[t][r] - mu) * rs * lng[nn] + lnb[nn];
      c[t][r] = fmaxf(z, 0.0f);
    }
  }
  // stage z tile to wave-private LDS
#pragma unroll
  for (int t = 0; t < 4; ++t)
#pragma unroll
    for (int r = 0; r < 8; ++r)
      zsh[wv][(r + (half << 3)) * 64 + (t << 4) + mrow] = c[t][r];
  __syncthreads();

  // final tiny GEMM: out[e, 0..9] = z[e,:] @ W2 + b2
  for (int o = lane; o < 160; o += 32) {
    int e = o / 10, cc = o - e * 10;
    if (e0 + e < E) {
      float acc = b2[cc];
      const float* zr = zsh[wv] + e * 64;
#pragma unroll
      for (int k = 0; k < 64; ++k) acc += zr[k] * W2[k * 10 + cc];
      out[(size_t)(e0 + e) * 10 + cc] = acc;
    }
  }
}

// ---------------- host ----------------

extern "C" void kernel_launch(void* const* d_in, const int* in_sizes, int n_in,
                              void* d_out, int out_size, void* d_ws, size_t ws_size,
                              hipStream_t stream) {
  const float* x     = (const float*)d_in[0];
  const float* W_lin = (const float*)d_in[1];
  const float* b_lin = (const float*)d_in[2];
  const float* Ws    = (const float*)d_in[3];
  const float* W1    = (const float*)d_in[4];
  const float* b1    = (const float*)d_in[5];
  const float* ln_g  = (const float*)d_in[6];
  const float* ln_b  = (const float*)d_in[7];
  const float* W2    = (const float*)d_in[8];
  const float* b2    = (const float*)d_in[9];
  const int*   ei    = (const int*)d_in[10];

  int N = in_sizes[0] / 64;
  int E = in_sizes[10] / 2;
  const int* row = ei;       // edge_index[0] = destinations for segment_sum
  const int* col = ei + E;   // edge_index[1] = gather sources

  char* ws = (char*)d_ws; (void)ws_size;
  size_t off = 0;
  auto alloc = [&](size_t bytes) {
    void* p = ws + off;
    off = (off + bytes + 255) & ~(size_t)255;
    return p;
  };
  float*    dinv   = (float*)alloc((size_t)N * 4);
  float*    hA     = (float*)alloc((size_t)N * 64 * 4);
  float*    hB     = (float*)alloc((size_t)N * 64 * 4);
  float*    h0     = (float*)alloc((size_t)N * 64 * 4);
  float*    agg    = (float*)alloc((size_t)N * 64 * 4);
  _Float16* Wlin_h = (_Float16*)alloc(64 * 64 * 2);
  _Float16* Ws_h   = (_Float16*)alloc(8 * 64 * 64 * 2);
  _Float16* W1_h   = (_Float16*)alloc(128 * 64 * 2);

  // f16 weight staging
  f32_to_f16_kernel<<<(4096 + 255) / 256, 256, 0, stream>>>(W_lin, Wlin_h, 4096);
  f32_to_f16_kernel<<<(32768 + 255) / 256, 256, 0, stream>>>(Ws, Ws_h, 32768);
  f32_to_f16_kernel<<<(8192 + 255) / 256, 256, 0, stream>>>(W1, W1_h, 8192);

  // gcn_norm
  hipMemsetAsync(dinv, 0, (size_t)N * 4, stream);
  deg_kernel<<<(E + 255) / 256, 256, 0, stream>>>(col, dinv, E);
  dinv_kernel<<<(N + 255) / 256, 256, 0, stream>>>(dinv, N);

  int node_tiles  = (N + 15) / 16;
  int node_blocks = (node_tiles + 7) / 8;   // 8 waves (256 threads) per block
  lin_kernel<<<node_blocks, 256, 0, stream>>>(x, Wlin_h, b_lin, hA, h0, N);

  float* cur = hA; float* nxt = hB;
  long long spmm_threads = (long long)E * 16;
  int spmm_blocks = (int)((spmm_threads + 255) / 256);
  for (int l = 0; l < 8; ++l) {
    float beta = logf(0.5f / (float)(l + 1) + 1.0f);
    hipMemsetAsync(agg, 0, (size_t)N * 64 * 4, stream);
    spmm_kernel<<<spmm_blocks, 256, 0, stream>>>(cur, row, col, dinv, agg, E);
    layer_kernel<<<node_blocks, 256, 0, stream>>>(agg, cur, h0, dinv,
                                                  Ws_h + (size_t)l * 4096, nxt, beta, N);
    float* t = cur; cur = nxt; nxt = t;
  }

  int edge_tiles  = (E + 15) / 16;
  int edge_blocks = (edge_tiles + 7) / 8;
  cls_kernel<<<edge_blocks, 256, 0, stream>>>(cur, row, col, W1_h, b1, ln_g, ln_b,
                                              W2, b2, (float*)d_out, E);
}
